// T2A_3384434229932
// MI455X (gfx1250) — matrix operand order repaired and checked by reading, not compiled, not run on hardware
//
#include <hip/hip_runtime.h>
#include <hip/hip_fp16.h>
#include <hip/hip_bf16.h>
#include <cstdint>
#include <cstddef>

// ---------------------------------------------------------------------------
// T5 encoder-decoder greedy decode for MI455X (gfx1250, wave32, WMMA).
//  * all GEMMs: V_WMMA_F32_16X16X32_F16, f32 accumulate, f16 operands
//  * each wave owns a 16x64 output tile: 4 independent accumulators ->
//    4 independent v_wmma per k-step (hides XDL latency), A fragment shared
//  * weights converted+transposed to f16 once per launch (~95MB -> fits L2)
//  * decoder runs incrementally with self-attn KV cache + hoisted cross K/V
//    (mathematically exact vs. the reference due to causal masking)
// ---------------------------------------------------------------------------

typedef _Float16 v16h __attribute__((ext_vector_type(16)));
typedef _Float16 v8h  __attribute__((ext_vector_type(8)));
typedef float    v8f  __attribute__((ext_vector_type(8)));

constexpr int B_   = 4;
constexpr int S_   = 64;
constexpr int D_   = 512;
constexpr int H_   = 8;
constexpr int DFF_ = 2048;
constexpr int L_   = 2;
constexpr int V_   = 32128;
constexpr int DH_  = 64;
constexpr int TOK  = B_ * S_;   // 256
constexpr int MAXT = 16;

// ---------------------------------------------------------------------------
// WMMA GEMM:  C[M,N](f32) = A[M,K](f16, row-major) * Bt[N,K](f16, row-major,
// i.e. original B[K,N] pre-transposed).  One wave per 16x64 tile (4 acc).
// EPI bit0: += resid ; bit1: relu ; bit2: also store f16 copy.
// Fragment addressing uses a contiguous k order for the 16x32 A fragment:
//   lanes 0-15 : row M=lane, halves 0..15 -> K+0..15
//   lanes 16-31: row M=lane-16, halves 0..15 -> K+16..31
// and the 32x16 B layout (lane%16 = N col; lanes<16 K+0..15, lanes>=16 K+16..31).
// ---------------------------------------------------------------------------
template <int EPI>
__global__ __launch_bounds__(32)
void k_gemm(const _Float16* __restrict__ A, const _Float16* __restrict__ Bt,
            float* __restrict__ C, _Float16* __restrict__ C16,
            const float* __restrict__ resid, int M, int N, int K)
{
    const int lane = threadIdx.x & 31;
    const int n0   = blockIdx.x * 64;
    const int m0   = blockIdx.y * 16;
    const int hi   = (lane < 16) ? 0 : 1;
    const int lr   = lane & 15;

    const _Float16* ap = A  + (size_t)(m0 + lr) * (size_t)K + (hi ? 16 : 0);
    const _Float16* bp = Bt + (size_t)(n0 + lr) * (size_t)K + (hi ? 16 : 0);
    const size_t tstep = (size_t)16 * (size_t)K;   // next N-tile of Bt

    v8f acc0 = {}, acc1 = {}, acc2 = {}, acc3 = {};
    for (int k = 0; k < K; k += 32) {
        __builtin_prefetch(bp + k + 256,              0, 3);  // WGP-scope prefetch
        __builtin_prefetch(bp + tstep     + k + 256,  0, 3);
        __builtin_prefetch(bp + tstep * 2 + k + 256,  0, 3);
        __builtin_prefetch(bp + tstep * 3 + k + 256,  0, 3);
        union { v16h v; v8h h[2]; } a, b0, b1, b2, b3;
        a.h[0]  = *(const v8h*)(ap + k);
        a.h[1]  = *(const v8h*)(ap + k + 8);
        b0.h[0] = *(const v8h*)(bp + k);
        b0.h[1] = *(const v8h*)(bp + k + 8);
        b1.h[0] = *(const v8h*)(bp + tstep + k);
        b1.h[1] = *(const v8h*)(bp + tstep + k + 8);
        b2.h[0] = *(const v8h*)(bp + tstep * 2 + k);
        b2.h[1] = *(const v8h*)(bp + tstep * 2 + k + 8);
        b3.h[0] = *(const v8h*)(bp + tstep * 3 + k);
        b3.h[1] = *(const v8h*)(bp + tstep * 3 + k + 8);
        acc0 = __builtin_amdgcn_wmma_f32_16x16x32_f16(false, a.v, false, b0.v, (short)0, acc0, false, false);
        acc1 = __builtin_amdgcn_wmma_f32_16x16x32_f16(false, a.v, false, b1.v, (short)0, acc1, false, false);
        acc2 = __builtin_amdgcn_wmma_f32_16x16x32_f16(false, a.v, false, b2.v, (short)0, acc2, false, false);
        acc3 = __builtin_amdgcn_wmma_f32_16x16x32_f16(false, a.v, false, b3.v, (short)0, acc3, false, false);
    }

    const int rb = m0 + (hi ? 8 : 0);
    v8f accs[4] = { acc0, acc1, acc2, acc3 };
#pragma unroll
    for (int t = 0; t < 4; ++t) {
        const int col = n0 + t * 16 + lr;
#pragma unroll
        for (int r = 0; r < 8; ++r) {
            size_t idx = (size_t)(rb + r) * (size_t)N + col;
            float v = accs[t][r];
            if (EPI & 1) v += resid[idx];
            if (EPI & 2) v = fmaxf(v, 0.0f);
            C[idx] = v;
            if (EPI & 4) C16[idx] = (_Float16)v;
        }
    }
}

// f32 -> f16 transpose: src[R,C] -> dst[C,R]
__global__ void k_transpose(const float* __restrict__ src,
                            _Float16* __restrict__ dst, int R, int C)
{
    int i = blockIdx.x * 256 + threadIdx.x;
    if (i >= R * C) return;
    int r = i / C, c = i % C;
    dst[(size_t)c * R + r] = (_Float16)src[i];
}

__global__ void k_zero_h(_Float16* p, int n)
{
    int i = blockIdx.x * 256 + threadIdx.x;
    if (i < n) p[i] = (_Float16)0.0f;
}

// embedding gather: x[tok,:] = emb[ids[tok],:]
__global__ void k_embed(const int* __restrict__ ids, const float* __restrict__ emb,
                        float* __restrict__ x)
{
    int tok = blockIdx.x, tid = threadIdx.x;
    int id = ids[tok];
    x[(size_t)tok * D_ + tid]       = emb[(size_t)id * D_ + tid];
    x[(size_t)tok * D_ + tid + 256] = emb[(size_t)id * D_ + tid + 256];
}

// encoder attention bias from mask: bias[b*S+s] = (1-mask)*NEG
__global__ void k_bias(const float* __restrict__ mask, float* __restrict__ bias)
{
    int i = threadIdx.x;           // 256 == B*S
    bias[i] = (1.0f - mask[i]) * -1e9f;
}

// decoder first token: x[r,:] = emb[PAD=0,:] for r<B else 0
__global__ void k_dec_init(const float* __restrict__ emb, float* __restrict__ x)
{
    int r = blockIdx.x, tid = threadIdx.x;
    float a = (r < B_) ? emb[tid]       : 0.0f;
    float b = (r < B_) ? emb[tid + 256] : 0.0f;
    x[(size_t)r * D_ + tid]       = a;
    x[(size_t)r * D_ + tid + 256] = b;
}

// T5 RMSNorm -> f16 output; rows >= nreal are zero-padded (for WMMA M=16 tiles)
__global__ void k_rmsnorm(const float* __restrict__ x, const float* __restrict__ w,
                          _Float16* __restrict__ out, int nreal)
{
    int row = blockIdx.x, tid = threadIdx.x;
    __shared__ float red[256];
    if (row >= nreal) {
        out[(size_t)row * D_ + tid]       = (_Float16)0.0f;
        out[(size_t)row * D_ + tid + 256] = (_Float16)0.0f;
        return;
    }
    const float* xr = x + (size_t)row * D_;
    float a = xr[tid], b = xr[tid + 256];
    red[tid] = a * a + b * b;
    __syncthreads();
    for (int s = 128; s > 0; s >>= 1) {
        if (tid < s) red[tid] += red[tid + s];
        __syncthreads();
    }
    float scale = rsqrtf(red[0] * (1.0f / (float)D_) + 1e-6f);
    out[(size_t)row * D_ + tid]       = (_Float16)(a * scale * w[tid]);
    out[(size_t)row * D_ + tid + 256] = (_Float16)(b * scale * w[tid + 256]);
}

// encoder self-attention (64 keys), grid (S,H,B), block 64
__global__ void k_attn_enc(const float* __restrict__ q, const float* __restrict__ k,
                           const float* __restrict__ v, const float* __restrict__ bias,
                           _Float16* __restrict__ out)
{
    int qi = blockIdx.x, h = blockIdx.y, b = blockIdx.z;
    int tid = threadIdx.x;                 // key index / out dim
    __shared__ float sm[64];
    const float* qrow = q + (size_t)(b * S_ + qi) * D_ + h * DH_;
    const float* krow = k + (size_t)(b * S_ + tid) * D_ + h * DH_;
    float s = 0.0f;
    for (int d = 0; d < DH_; ++d) s += qrow[d] * krow[d];
    s = s * 0.125f + bias[b * S_ + tid];
    sm[tid] = s;
    __syncthreads();
    float mx = -3.4e38f;
    for (int j = 0; j < S_; ++j) mx = fmaxf(mx, sm[j]);
    float e = __expf(s - mx);
    __syncthreads();
    sm[tid] = e;
    __syncthreads();
    float sum = 0.0f;
    for (int j = 0; j < S_; ++j) sum += sm[j];
    float inv = 1.0f / sum;
    const float* vb = v + (size_t)(b * S_) * D_ + h * DH_ + tid;
    float acc = 0.0f;
    for (int j = 0; j < S_; ++j) acc += sm[j] * vb[(size_t)j * D_];
    out[(size_t)(b * S_ + qi) * D_ + h * DH_ + tid] = (_Float16)(acc * inv);
}

// decoder cross-attention vs 64 encoder keys, grid (H, 16 rows), block 64
__global__ void k_attn_cross(const float* __restrict__ q, const float* __restrict__ kc,
                             const float* __restrict__ vc, const float* __restrict__ bias,
                             _Float16* __restrict__ out)
{
    int h = blockIdx.x, r = blockIdx.y, tid = threadIdx.x;
    if (r >= B_) { out[(size_t)r * D_ + h * DH_ + tid] = (_Float16)0.0f; return; }
    __shared__ float sm[64];
    const float* qrow = q + (size_t)r * D_ + h * DH_;
    const float* krow = kc + (size_t)(r * S_ + tid) * D_ + h * DH_;
    float s = 0.0f;
    for (int d = 0; d < DH_; ++d) s += qrow[d] * krow[d];
    s = s * 0.125f + bias[r * S_ + tid];
    sm[tid] = s;
    __syncthreads();
    float mx = -3.4e38f;
    for (int j = 0; j < S_; ++j) mx = fmaxf(mx, sm[j]);
    float e = __expf(s - mx);
    __syncthreads();
    sm[tid] = e;
    __syncthreads();
    float sum = 0.0f;
    for (int j = 0; j < S_; ++j) sum += sm[j];
    float inv = 1.0f / sum;
    const float* vb = vc + (size_t)(r * S_) * D_ + h * DH_ + tid;
    float acc = 0.0f;
    for (int j = 0; j < S_; ++j) acc += sm[j] * vb[(size_t)j * D_];
    out[(size_t)r * D_ + h * DH_ + tid] = (_Float16)(acc * inv);
}

// decoder self-attention over t cached keys, grid (H, 16 rows), block 64
__global__ void k_attn_self(const float* __restrict__ q, const float* __restrict__ kcache,
                            const float* __restrict__ vcache, int l, int t,
                            _Float16* __restrict__ out)
{
    int h = blockIdx.x, r = blockIdx.y, tid = threadIdx.x;
    if (r >= B_) { out[(size_t)r * D_ + h * DH_ + tid] = (_Float16)0.0f; return; }
    __shared__ float sm[64];
    const float* qrow = q + (size_t)r * D_ + h * DH_;
    if (tid < t) {
        const float* krow = kcache + (size_t)((l * B_ + r) * MAXT + tid) * D_ + h * DH_;
        float s = 0.0f;
        for (int d = 0; d < DH_; ++d) s += qrow[d] * krow[d];
        sm[tid] = s * 0.125f;
    }
    __syncthreads();
    float mx = -3.4e38f;
    for (int j = 0; j < t; ++j) mx = fmaxf(mx, sm[j]);
    __syncthreads();
    if (tid < t) sm[tid] = __expf(sm[tid] - mx);
    __syncthreads();
    float sum = 0.0f;
    for (int j = 0; j < t; ++j) sum += sm[j];
    float inv = 1.0f / sum;
    const float* vb = vcache + (size_t)((l * B_ + r) * MAXT) * D_ + h * DH_ + tid;
    float acc = 0.0f;
    for (int j = 0; j < t; ++j) acc += sm[j] * vb[(size_t)j * D_];
    out[(size_t)r * D_ + h * DH_ + tid] = (_Float16)(acc * inv);
}

// append this step's K/V (rows 0..B-1 of [16,D] buffers) into the caches
__global__ void k_kv_append(const float* __restrict__ kf, const float* __restrict__ vf,
                            float* __restrict__ kcache, float* __restrict__ vcache,
                            int l, int t)
{
    int b = blockIdx.x, d = threadIdx.x;   // blockDim = 512
    size_t dst = (size_t)((l * B_ + b) * MAXT + t) * D_ + d;
    kcache[dst] = kf[(size_t)b * D_ + d];
    vcache[dst] = vf[(size_t)b * D_ + d];
}

// softmax over V with argmax; writes prob_history (f32) + f16 probs + eos flag
__global__ void k_softmax_v(const float* __restrict__ logits, float* __restrict__ out_probs,
                            float* __restrict__ out_flags, _Float16* __restrict__ probs16,
                            int it)
{
    const int NT = 256;
    int b = blockIdx.x, tid = threadIdx.x;
    const float* row = logits + (size_t)b * V_;
    __shared__ float smax[NT];
    __shared__ int   sidx[NT];
    __shared__ float ssum[NT];
    float mx = -3.4e38f; int mi = 0;
    for (int v = tid; v < V_; v += NT) {
        float x = row[v];
        if (x > mx) { mx = x; mi = v; }
    }
    smax[tid] = mx; sidx[tid] = mi;
    __syncthreads();
    for (int s = NT / 2; s > 0; s >>= 1) {
        if (tid < s && smax[tid + s] > smax[tid]) {
            smax[tid] = smax[tid + s]; sidx[tid] = sidx[tid + s];
        }
        __syncthreads();
    }
    float gmax = smax[0]; int gidx = sidx[0];
    __syncthreads();
    float* prow = out_probs + (size_t)(b * MAXT + it) * V_;
    float lsum = 0.0f;
    for (int v = tid; v < V_; v += NT) {
        float e = __expf(row[v] - gmax);
        prow[v] = e; lsum += e;
    }
    ssum[tid] = lsum;
    __syncthreads();
    for (int s = NT / 2; s > 0; s >>= 1) {
        if (tid < s) ssum[tid] += ssum[tid + s];
        __syncthreads();
    }
    float inv = 1.0f / ssum[0];
    _Float16* p16 = probs16 + (size_t)b * V_;
    for (int v = tid; v < V_; v += NT) {
        float p = prow[v] * inv;
        prow[v] = p;
        p16[v] = (_Float16)p;
    }
    if (tid == 0) out_flags[b * MAXT + it] = (gidx == 0) ? 1.0f : 0.0f;  // PAD_ID==0
}

// ---------------------------------------------------------------------------
// host helpers
// ---------------------------------------------------------------------------
static void gemm(hipStream_t st, const _Float16* A, const _Float16* Bt, float* C,
                 _Float16* C16, const float* resid, int M, int N, int K, int epi)
{
    dim3 g(N / 64, M / 16), b(32);
    switch (epi) {
    case 0: hipLaunchKernelGGL(HIP_KERNEL_NAME(k_gemm<0>), g, b, 0, st, A, Bt, C, C16, resid, M, N, K); break;
    case 1: hipLaunchKernelGGL(HIP_KERNEL_NAME(k_gemm<1>), g, b, 0, st, A, Bt, C, C16, resid, M, N, K); break;
    case 6: hipLaunchKernelGGL(HIP_KERNEL_NAME(k_gemm<6>), g, b, 0, st, A, Bt, C, C16, resid, M, N, K); break;
    }
}

static void tr(hipStream_t st, const float* src, _Float16* dst, int R, int C)
{
    int total = R * C;
    k_transpose<<<(total + 255) / 256, 256, 0, st>>>(src, dst, R, C);
}

extern "C" void kernel_launch(void* const* d_in, const int* in_sizes, int n_in,
                              void* d_out, int out_size, void* d_ws, size_t ws_size,
                              hipStream_t stream)
{
    (void)in_sizes; (void)n_in; (void)out_size;

    const int*   ids   = (const int*)  d_in[0];
    const float* mask  = (const float*)d_in[1];
    const float* emb   = (const float*)d_in[2];
    const float* e_wq  = (const float*)d_in[3];
    const float* e_wk  = (const float*)d_in[4];
    const float* e_wv  = (const float*)d_in[5];
    const float* e_wo  = (const float*)d_in[6];
    const float* e_ln1 = (const float*)d_in[7];
    const float* e_w1  = (const float*)d_in[8];
    const float* e_w2  = (const float*)d_in[9];
    const float* e_ln2 = (const float*)d_in[10];
    const float* e_lnf = (const float*)d_in[11];
    const float* d_sq  = (const float*)d_in[12];
    const float* d_sk  = (const float*)d_in[13];
    const float* d_sv  = (const float*)d_in[14];
    const float* d_so  = (const float*)d_in[15];
    const float* d_ln1 = (const float*)d_in[16];
    const float* d_cq  = (const float*)d_in[17];
    const float* d_ck  = (const float*)d_in[18];
    const float* d_cv  = (const float*)d_in[19];
    const float* d_co  = (const float*)d_in[20];
    const float* d_ln2 = (const float*)d_in[21];
    const float* d_w1  = (const float*)d_in[22];
    const float* d_w2  = (const float*)d_in[23];
    const float* d_ln3 = (const float*)d_in[24];
    const float* d_lnf = (const float*)d_in[25];
    const float* lm    = (const float*)d_in[26];

    float* out_probs = (float*)d_out;
    float* out_flags = out_probs + (size_t)B_ * MAXT * V_;

    // ---- deterministic workspace carve (needs ~106 MB) ----
    char* base = (char*)d_ws;
    char* p = base;
    auto take = [&](size_t bytes) -> char* {
        char* r = p;
        p += (bytes + 255) & ~(size_t)255;
        return r;
    };
    const size_t DD  = (size_t)L_ * D_ * D_;     // per-tensor f16 elems (proj)
    const size_t DF  = (size_t)L_ * D_ * DFF_;
    _Float16* eWqT = (_Float16*)take(DD * 2);
    _Float16* eWkT = (_Float16*)take(DD * 2);
    _Float16* eWvT = (_Float16*)take(DD * 2);
    _Float16* eWoT = (_Float16*)take(DD * 2);
    _Float16* eW1T = (_Float16*)take(DF * 2);
    _Float16* eW2T = (_Float16*)take(DF * 2);
    _Float16* dSqT = (_Float16*)take(DD * 2);
    _Float16* dSkT = (_Float16*)take(DD * 2);
    _Float16* dSvT = (_Float16*)take(DD * 2);
    _Float16* dSoT = (_Float16*)take(DD * 2);
    _Float16* dCqT = (_Float16*)take(DD * 2);
    _Float16* dCkT = (_Float16*)take(DD * 2);
    _Float16* dCvT = (_Float16*)take(DD * 2);
    _Float16* dCoT = (_Float16*)take(DD * 2);
    _Float16* dW1T = (_Float16*)take(DF * 2);
    _Float16* dW2T = (_Float16*)take(DF * 2);
    _Float16* embT = (_Float16*)take((size_t)V_ * D_ * 2);
    _Float16* lmT  = (_Float16*)take((size_t)V_ * D_ * 2);
    _Float16* h16e = (_Float16*)take((size_t)TOK * D_ * 2);
    _Float16* ao16e= (_Float16*)take((size_t)TOK * D_ * 2);
    _Float16* ff16 = (_Float16*)take((size_t)TOK * DFF_ * 2);
    _Float16* hs16 = (_Float16*)take((size_t)TOK * D_ * 2);
    _Float16* h16d = (_Float16*)take((size_t)MAXT * D_ * 2);
    _Float16* ao16d= (_Float16*)take((size_t)MAXT * D_ * 2);
    _Float16* dff16= (_Float16*)take((size_t)MAXT * DFF_ * 2);
    _Float16* d16  = (_Float16*)take((size_t)MAXT * D_ * 2);
    _Float16* probs16 = (_Float16*)take((size_t)MAXT * V_ * 2);
    float* xe    = (float*)take((size_t)TOK * D_ * 4);
    float* qf    = (float*)take((size_t)TOK * D_ * 4);
    float* kf    = (float*)take((size_t)TOK * D_ * 4);
    float* vf    = (float*)take((size_t)TOK * D_ * 4);
    float* big32 = (float*)take((size_t)TOK * DFF_ * 4);     // FFN f32 / logits[16,V]
    float* ckvK  = (float*)take((size_t)L_ * TOK * D_ * 4);
    float* ckvV  = (float*)take((size_t)L_ * TOK * D_ * 4);
    float* xd    = (float*)take((size_t)MAXT * D_ * 4);
    float* kcache= (float*)take((size_t)L_ * B_ * MAXT * D_ * 4);
    float* vcache= (float*)take((size_t)L_ * B_ * MAXT * D_ * 4);
    float* bias  = (float*)take((size_t)B_ * S_ * 4);
    if ((size_t)(p - base) > ws_size) return;   // workspace too small: bail

    // ---- one-time (per launch) weight convert + transpose to f16 [N,K] ----
    for (int l = 0; l < L_; ++l) {
        size_t o1 = (size_t)l * D_ * D_, o2 = (size_t)l * D_ * DFF_;
        tr(stream, e_wq + o1, eWqT + o1, D_, D_);
        tr(stream, e_wk + o1, eWkT + o1, D_, D_);
        tr(stream, e_wv + o1, eWvT + o1, D_, D_);
        tr(stream, e_wo + o1, eWoT + o1, D_, D_);
        tr(stream, e_w1 + o2, eW1T + o2, D_, DFF_);
        tr(stream, e_w2 + o2, eW2T + o2, DFF_, D_);
        tr(stream, d_sq + o1, dSqT + o1, D_, D_);
        tr(stream, d_sk + o1, dSkT + o1, D_, D_);
        tr(stream, d_sv + o1, dSvT + o1, D_, D_);
        tr(stream, d_so + o1, dSoT + o1, D_, D_);
        tr(stream, d_cq + o1, dCqT + o1, D_, D_);
        tr(stream, d_ck + o1, dCkT + o1, D_, D_);
        tr(stream, d_cv + o1, dCvT + o1, D_, D_);
        tr(stream, d_co + o1, dCoT + o1, D_, D_);
        tr(stream, d_w1 + o2, dW1T + o2, D_, DFF_);
        tr(stream, d_w2 + o2, dW2T + o2, DFF_, D_);
    }
    tr(stream, emb, embT, V_, D_);   // embT [D,V]
    tr(stream, lm,  lmT,  D_, V_);   // lmT  [V,D]

    // ---- encoder ----
    k_embed<<<TOK, 256, 0, stream>>>(ids, emb, xe);
    k_bias<<<1, 256, 0, stream>>>(mask, bias);
    for (int l = 0; l < L_; ++l) {
        size_t o1 = (size_t)l * D_ * D_, o2 = (size_t)l * D_ * DFF_;
        k_rmsnorm<<<TOK, 256, 0, stream>>>(xe, e_ln1 + l * D_, h16e, TOK);
        gemm(stream, h16e, eWqT + o1, qf, nullptr, nullptr, TOK, D_, D_, 0);
        gemm(stream, h16e, eWkT + o1, kf, nullptr, nullptr, TOK, D_, D_, 0);
        gemm(stream, h16e, eWvT + o1, vf, nullptr, nullptr, TOK, D_, D_, 0);
        k_attn_enc<<<dim3(S_, H_, B_), 64, 0, stream>>>(qf, kf, vf, bias, ao16e);
        gemm(stream, ao16e, eWoT + o1, xe, nullptr, xe, TOK, D_, D_, 1);
        k_rmsnorm<<<TOK, 256, 0, stream>>>(xe, e_ln2 + l * D_, h16e, TOK);
        gemm(stream, h16e, eW1T + o2, big32, ff16, nullptr, TOK, DFF_, D_, 6);
        gemm(stream, ff16, eW2T + o2, xe, nullptr, xe, TOK, D_, DFF_, 1);
    }
    k_rmsnorm<<<TOK, 256, 0, stream>>>(xe, e_lnf, hs16, TOK);

    // hoisted cross-attention K/V (invariant across decode steps)
    for (int l = 0; l < L_; ++l) {
        size_t o1 = (size_t)l * D_ * D_;
        gemm(stream, hs16, dCkT + o1, ckvK + (size_t)l * TOK * D_, nullptr, nullptr, TOK, D_, D_, 0);
        gemm(stream, hs16, dCvT + o1, ckvV + (size_t)l * TOK * D_, nullptr, nullptr, TOK, D_, D_, 0);
    }
    { // zero pad rows of probs16 once (rows B..15 of the M=16 WMMA tile)
        int n = (MAXT - B_) * V_;
        k_zero_h<<<(n + 255) / 256, 256, 0, stream>>>(probs16 + (size_t)B_ * V_, n);
    }

    // ---- incremental decoder, 16 steps ----
    for (int it = 0; it < MAXT; ++it) {
        if (it == 0)
            k_dec_init<<<MAXT, 256, 0, stream>>>(emb, xd);
        else  // y_t = probs @ emb  (soft embedding lookup)
            gemm(stream, probs16, embT, xd, nullptr, nullptr, MAXT, D_, V_, 0);

        for (int l = 0; l < L_; ++l) {
            size_t o1 = (size_t)l * D_ * D_, o2 = (size_t)l * D_ * DFF_;
            k_rmsnorm<<<MAXT, 256, 0, stream>>>(xd, d_ln1 + l * D_, h16d, B_);
            gemm(stream, h16d, dSqT + o1, qf, nullptr, nullptr, MAXT, D_, D_, 0);
            gemm(stream, h16d, dSkT + o1, kf, nullptr, nullptr, MAXT, D_, D_, 0);
            gemm(stream, h16d, dSvT + o1, vf, nullptr, nullptr, MAXT, D_, D_, 0);
            k_kv_append<<<B_, D_, 0, stream>>>(kf, vf, kcache, vcache, l, it);
            k_attn_self<<<dim3(H_, MAXT), 64, 0, stream>>>(qf, kcache, vcache, l, it + 1, ao16d);
            gemm(stream, ao16d, dSoT + o1, xd, nullptr, xd, MAXT, D_, D_, 1);

            k_rmsnorm<<<MAXT, 256, 0, stream>>>(xd, d_ln2 + l * D_, h16d, B_);
            gemm(stream, h16d, dCqT + o1, qf, nullptr, nullptr, MAXT, D_, D_, 0);
            k_attn_cross<<<dim3(H_, MAXT), 64, 0, stream>>>(
                qf, ckvK + (size_t)l * TOK * D_, ckvV + (size_t)l * TOK * D_, bias, ao16d);
            gemm(stream, ao16d, dCoT + o1, xd, nullptr, xd, MAXT, D_, D_, 1);

            k_rmsnorm<<<MAXT, 256, 0, stream>>>(xd, d_ln3 + l * D_, h16d, B_);
            gemm(stream, h16d, dW1T + o2, big32, dff16, nullptr, MAXT, DFF_, D_, 6);
            gemm(stream, dff16, dW2T + o2, xd, nullptr, xd, MAXT, D_, DFF_, 1);
        }
        k_rmsnorm<<<MAXT, 256, 0, stream>>>(xd, d_lnf, d16, B_);
        gemm(stream, d16, lmT, big32, nullptr, nullptr, MAXT, V_, D_, 0);  // logits [16,V]
        k_softmax_v<<<B_, 256, 0, stream>>>(big32, out_probs, out_flags, probs16, it);
    }
}
